// HarmonicsExcitation_35785667510965
// MI455X (gfx1250) — compile-verified, matrix operand order checked by reading, not run on hardware
//
#include <hip/hip_runtime.h>

// ---------------------------------------------------------------------------
// HarmonicsExcitation on gfx1250 (MI455X, wave32, WMMA)
//
// Pipeline:
//   K1 scan_kernel : closed-form per-frame phase sums + f64 scan -> phase0[16384]
//   K2 bfrag_kernel: build fixed 1024x512 "istft(truncated stft)" operator K(2n-m)
//                    directly in WMMA B-fragment layout, f16 hi/lo split
//   K3 synth_kernel: per-sample phase (closed form) + band masks + wavetable lerp
//                    -> p2x[7,864,320] f32
//   K4 gemm_kernel : frames(16385x1024, hann1024-windowed, reflect pad) x K
//                    via v_wmma_f32_16x16x32_f16, 3-product f16 split (~f32 acc).
//                    A split precomputed once into LDS (packed u32 halves);
//                    each wave drives 4 accumulators (4 N-tiles) per A fragment.
//   K5 ola_kernel  : overlap-add (hop 240) + hann^2 envelope divide -> out
// ---------------------------------------------------------------------------

#define L2X      7864320L     // p_out2x length (16384*480)
#define NFRAMES  16384        // f0 frames
#define NSTFT    16385        // stft frames
#define MTILES   1025         // ceil(16385/16)
#define OUTLEN   3932160L     // (16385-1)*240
#define PI_D     3.14159265358979323846

typedef _Float16 half_t;
typedef _Float16 v16h __attribute__((ext_vector_type(16)));
typedef float    v8f  __attribute__((ext_vector_type(8)));
typedef unsigned int u32;

// ---- closed-form per-frame phase increment -------------------------------
__device__ __forceinline__ double frame_sum(const float* __restrict__ f0, int i) {
  float fi = f0[i];
  float fn = (i + 1 < NFRAMES) ? f0[i + 1] : 0.0f;
  double a = (fi != 0.0f) ? (double)fi : (double)fn;
  double b = (fn != 0.0f) ? (double)fn : (double)fi;
  // sum_{j=0..479} (a + (b-a)*j/480) / 96000
  return (480.0 * a + 239.5 * (b - a)) * (1.0 / 96000.0);
}

__global__ void scan_kernel(const float* __restrict__ f0, double* __restrict__ phase0) {
  __shared__ double chunk[256];
  int tid = threadIdx.x;
  int base = tid * 64;                       // 256 threads * 64 frames = 16384
  double s = 0.0;
  for (int j = 0; j < 64; ++j) s += frame_sum(f0, base + j);
  chunk[tid] = s;
  __syncthreads();
  if (tid == 0) {
    double acc = 0.0;
    for (int t = 0; t < 256; ++t) { double c = chunk[t]; chunk[t] = acc; acc = fmod(acc + c, 1.0); }
  }
  __syncthreads();
  double acc = chunk[tid];
  for (int j = 0; j < 64; ++j) {
    int i = base + j;
    phase0[i] = acc;                         // exclusive prefix, mod 1
    acc = fmod(acc + frame_sum(f0, i), 1.0);
  }
}

// ---- B operator in WMMA fragment layout ----------------------------------
// B[m][n] = K(2n - m), K(d) = (2*S - 1 - cos(256*th))/512, th = pi*d/512,
// S = sin(128.5*th)*cos(128*th)/sin(0.5*th); K(0) = 1.
// Fragment layout (16x16x32 f16 B, wave32): lane l -> column n = l&15,
// half h -> k_local = (h>>3)*16 + ((l>>4)*8) + (h&7).
__global__ void bfrag_kernel(u32* __restrict__ Bhi, u32* __restrict__ Blo) {
  int idx = blockIdx.x * blockDim.x + threadIdx.x;    // kt*1024 + tn*32 + lane
  if (idx >= 32 * 32 * 32) return;
  int lane = idx & 31;
  int tn   = (idx >> 5) & 31;
  int kt   = idx >> 10;
  int n     = tn * 16 + (lane & 15);
  int koff8 = (lane >> 4) * 8;
  u32 uh[8], ul[8];
#pragma unroll
  for (int g = 0; g < 8; ++g) { uh[g] = 0u; ul[g] = 0u; }
#pragma unroll
  for (int h = 0; h < 16; ++h) {
    int kl = ((h >> 3) * 16) + koff8 + (h & 7);
    int m  = kt * 32 + kl;
    int d  = 2 * n - m;
    double val;
    if (d == 0) {
      val = 1.0;
    } else {
      double th = PI_D * (double)d * (1.0 / 512.0);
      double S  = sin(128.5 * th) * cos(128.0 * th) / sin(0.5 * th);
      val = (2.0 * S - 1.0 - cos(256.0 * th)) * (1.0 / 512.0);
    }
    float  f  = (float)val;
    half_t hi = (half_t)f;
    half_t lo = (half_t)(f - (float)hi);
    unsigned short hu = __builtin_bit_cast(unsigned short, hi);
    unsigned short lu = __builtin_bit_cast(unsigned short, lo);
    int g = h >> 1, sh = (h & 1) * 16;
    uh[g] |= ((u32)hu) << sh;
    ul[g] |= ((u32)lu) << sh;
  }
  long o = (long)idx * 8;
#pragma unroll
  for (int g = 0; g < 8; ++g) { Bhi[o + g] = uh[g]; Blo[o + g] = ul[g]; }
}

// ---- synthesis ------------------------------------------------------------
__device__ __forceinline__ float table_lerp(const float* __restrict__ py, int S, float phase) {
  float mi  = (float)(S - 1);
  float idx = phase * mi;
  if (idx < 0.0f) idx = 0.0f;
  if (idx > mi)   idx = mi;
  int lo = (int)idx;
  if (lo > S - 2) lo = S - 2;
  float t = idx - (float)lo;
  float a = py[lo], b = py[lo + 1];
  return a + (b - a) * t;
}

__device__ __forceinline__ int inband(float f, double lo, double hi) {
  double d = (double)f;
  return (d >= lo) && (d < hi);
}

__global__ void synth_kernel(const float* __restrict__ f0,
                             const double* __restrict__ phase0,
                             const float* __restrict__ py48,  int S48,
                             const float* __restrict__ py144, int S144,
                             const float* __restrict__ py432, int S432,
                             float* __restrict__ p2x) {
  long s = (long)blockIdx.x * blockDim.x + threadIdx.x;
  if (s >= L2X) return;
  int i = (int)(s / 480);
  int j = (int)(s - (long)i * 480);
  float fi = f0[i];
  float fn = (i + 1 < NFRAMES) ? f0[i + 1] : 0.0f;
  double a = (fi != 0.0f) ? (double)fi : (double)fn;
  double b = (fn != 0.0f) ? (double)fn : (double)fi;
  // inclusive in-frame cumsum at sample j (closed form)
  double cumj = ((double)(j + 1) * a +
                 (b - a) * ((double)(j * (j + 1)) * (1.0 / 960.0))) * (1.0 / 96000.0);
  float phase = (float)fmod(phase0[i] + cumj, 1.0);
  double xs = (double)j * (1.0 / 480.0);
  double sn = sin(xs * (PI_D * 0.5));
  float tr = (float)(sn * sn);               // trans2x[j]
  const double B0 = 48000.0 / 2.0 / 16.0;    // 1500
  const double B1 = 48000.0 / 2.0 / 48.0;    // 500
  const double B2 = 48000.0 / 2.0 / 144.0;
  const double B3 = 48000.0 / 2.0 / 432.0;
  float out = 0.0f;
  {
    int v0 = inband(fi, B1, B0), v1 = inband(fn, B1, B0);
    if (v0 | v1) {
      float mask = v0 ? (v1 ? 1.0f : 1.0f - tr) : tr;
      out += mask * table_lerp(py48, S48, phase);
    }
  }
  {
    int v0 = inband(fi, B2, B1), v1 = inband(fn, B2, B1);
    if (v0 | v1) {
      float mask = v0 ? (v1 ? 1.0f : 1.0f - tr) : tr;
      out += mask * table_lerp(py144, S144, phase);
    }
  }
  {
    int v0 = inband(fi, B3, B2), v1 = inband(fn, B3, B2);
    if (v0 | v1) {
      float mask = v0 ? (v1 ? 1.0f : 1.0f - tr) : tr;
      out += mask * table_lerp(py432, S432, phase);
    }
  }
  float f0c = (float)(a + (b - a) * xs);
  p2x[s] = out * f0c * (float)(1.0 / 48000.0);
}

// ---- WMMA GEMM: ys[t][n] = hann512[n] * sum_m xw[t][m] * B[m][n] ----------
// A tile staged in LDS already f16-split and pair-packed: one u32 holds the
// halves of two adjacent columns == one VGPR of the v16h A fragment.
// Fragment build is therefore 2x ds_load_b128 per matrix, no VALU.
#define AS32 260   // u32 row stride (multiple of 4 -> 16B-aligned b128 reads)

__global__ void __launch_bounds__(128)
gemm_kernel(const float* __restrict__ p2x,
            const u32* __restrict__ Bhi, const u32* __restrict__ Blo,
            float* __restrict__ ys) {
  __shared__ u32 AhiL[16 * AS32];            // 16,640 B
  __shared__ u32 AloL[16 * AS32];            // 16,640 B
  int tm   = blockIdx.y;
  int tid  = threadIdx.x;
  int wave = tid >> 5;
  int lane = tid & 31;
  int tng  = (blockIdx.x * 4 + wave) * 4;    // first of 4 N-tiles for this wave
  int arow = lane & 15;
  int c0base = (lane >> 4) * 4;              // u32 offset: lanes>=16 -> cols +8
  const float c1024 = (float)(2.0 * PI_D / 1024.0);
  const float c512  = (float)(2.0 * PI_D / 512.0);

  v8f acc[4];
#pragma unroll
  for (int u = 0; u < 4; ++u) acc[u] = (v8f){0.f,0.f,0.f,0.f,0.f,0.f,0.f,0.f};

  for (int phse = 0; phse < 2; ++phse) {     // K split into two 512-wide halves
    __syncthreads();
    int mbase = phse * 512;
    // stage + window + f16-split A tile: 16 rows x 256 column-pairs
    for (int e = tid; e < 16 * 256; e += 128) {
      int row = e >> 8;
      int pc  = e & 255;                     // pair-column
      int m0  = mbase + pc * 2;
      long t  = (long)tm * 16 + row;
      float v0 = 0.0f, v1 = 0.0f;
      if (t < NSTFT) {
        long g0 = t * 480 + (long)m0 - 512;  // reflect-padded index
        long g1 = g0 + 1;
        if (g0 < 0) g0 = -g0;
        if (g0 >= L2X) g0 = 2 * L2X - 2 - g0;
        if (g1 < 0) g1 = -g1;
        if (g1 >= L2X) g1 = 2 * L2X - 2 - g1;
        float w0 = 0.5f - 0.5f * __cosf(c1024 * (float)m0);        // hann1024
        float w1 = 0.5f - 0.5f * __cosf(c1024 * (float)(m0 + 1));
        v0 = p2x[g0] * w0;
        v1 = p2x[g1] * w1;
      }
      half_t h0 = (half_t)v0;
      half_t h1 = (half_t)v1;
      half_t l0 = (half_t)(v0 - (float)h0);
      half_t l1 = (half_t)(v1 - (float)h1);
      u32 hp = (u32)__builtin_bit_cast(unsigned short, h0) |
               ((u32)__builtin_bit_cast(unsigned short, h1) << 16);
      u32 lp = (u32)__builtin_bit_cast(unsigned short, l0) |
               ((u32)__builtin_bit_cast(unsigned short, l1) << 16);
      AhiL[row * AS32 + pc] = hp;
      AloL[row * AS32 + pc] = lp;
    }
    __syncthreads();

    for (int kt2 = 0; kt2 < 16; ++kt2) {
      int kt = phse * 16 + kt2;
      int c0 = kt2 * 16 + c0base;
      union FU { uint4 q[2]; v16h h; };
      FU ah, al;
      const u32* hr = &AhiL[arow * AS32 + c0];
      const u32* lr = &AloL[arow * AS32 + c0];
      ah.q[0] = *(const uint4*)(hr);         // halves 0..7  (cols base..base+7)
      ah.q[1] = *(const uint4*)(hr + 8);     // halves 8..15 (cols base+16..23)
      al.q[0] = *(const uint4*)(lr);
      al.q[1] = *(const uint4*)(lr + 8);
#pragma unroll
      for (int u = 0; u < 4; ++u) {
        int tn = tng + u;
        long bo = (((long)kt * 32 + tn) * 32 + lane) * 8;
        FU bh, bl;
        const uint4* bhp = (const uint4*)(Bhi + bo);
        const uint4* blp = (const uint4*)(Blo + bo);
        bh.q[0] = bhp[0]; bh.q[1] = bhp[1];
        bl.q[0] = blp[0]; bl.q[1] = blp[1];
        // 3-product f16 split: ~f32 accuracy, f32 accumulate
        acc[u] = __builtin_amdgcn_wmma_f32_16x16x32_f16(false, ah.h, false, bh.h, (short)0, acc[u], false, false);
        acc[u] = __builtin_amdgcn_wmma_f32_16x16x32_f16(false, ah.h, false, bl.h, (short)0, acc[u], false, false);
        acc[u] = __builtin_amdgcn_wmma_f32_16x16x32_f16(false, al.h, false, bh.h, (short)0, acc[u], false, false);
      }
    }
  }

  int m0row = (lane >> 4) * 8;               // D rows: vgpr r -> M = r + m0row
#pragma unroll
  for (int u = 0; u < 4; ++u) {
    int n = (tng + u) * 16 + (lane & 15);
    float w512 = 0.5f - 0.5f * __cosf(c512 * (float)n);            // hann512
#pragma unroll
    for (int r = 0; r < 8; ++r) {
      long t = (long)tm * 16 + m0row + r;
      if (t < NSTFT) ys[t * 512 + n] = acc[u][r] * w512;
    }
  }
}

// ---- overlap-add + envelope normalize -------------------------------------
__global__ void ola_kernel(const float* __restrict__ ys, float* __restrict__ out) {
  long p = (long)blockIdx.x * blockDim.x + threadIdx.x;
  if (p >= OUTLEN) return;
  long q = p + 256;                           // istft pad = 512/2
  long t0 = (q - 272) / 240;  if (t0 < 0) t0 = 0;          // ceil((q-511)/240)
  long t1 = q / 240;          if (t1 > NSTFT - 1) t1 = NSTFT - 1;
  const float c512 = (float)(2.0 * PI_D / 512.0);
  float sig = 0.0f, env = 0.0f;
  for (long t = t0; t <= t1; ++t) {
    int nn = (int)(q - t * 240);              // 0..511
    float w = 0.5f - 0.5f * __cosf(c512 * (float)nn);
    sig += ys[t * 512 + nn];                  // ys already includes one window
    env += w * w;
  }
  out[p] = (env > 1e-11f) ? (sig / env) : 0.0f;
}

// ---------------------------------------------------------------------------
extern "C" void kernel_launch(void* const* d_in, const int* in_sizes, int n_in,
                              void* d_out, int out_size, void* d_ws, size_t ws_size,
                              hipStream_t stream) {
  const float* f0    = (const float*)d_in[0];
  const float* py48  = (const float*)d_in[1];
  const float* py144 = (const float*)d_in[2];
  const float* py432 = (const float*)d_in[3];
  float* out = (float*)d_out;

  char* ws = (char*)d_ws;
  size_t o = 0;
  float*  p2x    = (float*)(ws + o);  o += (((size_t)L2X * 4) + 255) & ~(size_t)255;
  double* phase0 = (double*)(ws + o); o += (size_t)NFRAMES * 8;
  u32*    Bhi    = (u32*)(ws + o);    o += (size_t)32 * 32 * 32 * 8 * 4;  // 1 MB
  u32*    Blo    = (u32*)(ws + o);    o += (size_t)32 * 32 * 32 * 8 * 4;  // 1 MB
  float*  ys     = (float*)(ws + o);  o += (size_t)MTILES * 16 * 512 * 4; // ~34 MB

  scan_kernel<<<1, 256, 0, stream>>>(f0, phase0);
  bfrag_kernel<<<(32 * 32 * 32 + 255) / 256, 256, 0, stream>>>(Bhi, Blo);
  synth_kernel<<<(int)((L2X + 255) / 256), 256, 0, stream>>>(
      f0, phase0, py48, in_sizes[1], py144, in_sizes[2], py432, in_sizes[3], p2x);
  gemm_kernel<<<dim3(2, MTILES), 128, 0, stream>>>(p2x, Bhi, Blo, ys);
  ola_kernel<<<(int)((OUTLEN + 255) / 256), 256, 0, stream>>>(ys, out);
}
